// DeformAtt_26371099198376
// MI455X (gfx1250) — matrix operand order repaired and checked by reading, not compiled, not run on hardware
//
#include <hip/hip_runtime.h>
#include <hip/hip_bf16.h>
#include <math.h>

// ---------------------------------------------------------------------------
// Problem constants (B=8, C=256, H=W=64, K=3)
// Reduction ordering for the 3x3 convs is TAP-MAJOR:  k = kk*256 + c.
// A 32-wide k-chunk then has constant kk and contiguous channels.
// ---------------------------------------------------------------------------
#define BATCH   8
#define CDIM    256
#define HDIM    64
#define WDIM    64
#define HW      4096            // H*W
#define NPIX    (BATCH * HW)    // 32768
#define KK      9               // 3x3 taps
#define KRED    (CDIM * KK)     // 2304
#define KT_REG  (KRED / 32)     // 72 k-chunks of 32
#define NT_REG  (CDIM / 16)     // 16 n-tiles of 16
#define KT_MASK (CDIM / 32)     // 8 k-chunks for the 1x1 conv
#define TILES_PER_BATCH (HW / 16)   // 256
#define NTILES  (NPIX / 16)         // 2048

typedef __attribute__((ext_vector_type(16))) _Float16 v16h;
typedef __attribute__((ext_vector_type(8)))  _Float16 v8h;
typedef __attribute__((ext_vector_type(2)))  _Float16 v2h;
typedef __attribute__((ext_vector_type(8)))  float    v8f;

__device__ __forceinline__ v8f wmma_f16(v16h a, v16h b, v8f c) {
    // D = A(16x32 f16) * B(32x16 f16) + C(16x16 f32)
    return __builtin_amdgcn_wmma_f32_16x16x32_f16(
        /*neg_a=*/false, a, /*neg_b=*/false, b,
        /*c_mod=*/(short)0, c, /*reuse_a=*/false, /*reuse_b=*/false);
}

// A-fragment (16x32 f16) from an LDS tile stored row-major, stride 40 halves.
// ISA A-layout: lane%16 = row M; lanes 0-15 hold k {0..7,16..23}, lanes 16-31
// hold k {8..15,24..31}. Two 16-byte LDS vector loads per lane.
__device__ __forceinline__ v16h load_afrag_lds(const _Float16* sA, int lane) {
    int row = lane & 15;
    int kb  = (lane >> 4) * 8;
    v8h lo = *(const v8h*)(sA + row * 40 + kb);
    v8h hi = *(const v8h*)(sA + row * 40 + kb + 16);
    v16h a;
#pragma unroll
    for (int i = 0; i < 8; ++i) { a[i] = lo[i]; a[i + 8] = hi[i]; }
    return a;
}

// Pre-packed B fragments: flat = ((kt*NT + nt)*32 + lane)*16 + e,
// with k = kt*32 + (lane>>4)*8 + (e&7) + (e>>3)*16,  n = nt*16 + (lane&15).
__device__ __forceinline__ v16h load_bfrag(const _Float16* Bh, int kt, int NT,
                                           int nt, int lane) {
    return *(const v16h*)(Bh + (((size_t)(kt * NT + nt) * 32 + lane) << 4));
}

// ---------------------------------------------------------------------------
// Prep kernels
// ---------------------------------------------------------------------------

// x NCHW f32 -> xh NHWC f16.  One block per (b,c) row: coalesced reads.
__global__ void prep_xh_kernel(const float* __restrict__ x,
                               _Float16* __restrict__ xh) {
    int row = blockIdx.x;                  // b*C + c
    int b = row >> 8, c = row & 255;
    const float* src = x + (size_t)row * HW;
    for (int hw = threadIdx.x; hw < HW; hw += blockDim.x)
        xh[((size_t)(b * HW + hw) * CDIM) + c] = (_Float16)src[hw];
}

// Pack w_reg [O=256][C=256][3][3] f32 into B fragments, tap-major k.
__global__ void prep_breg_kernel(const float* __restrict__ w_reg,
                                 _Float16* __restrict__ Bh) {
    int id = blockIdx.x * blockDim.x + threadIdx.x;   // 72*16*32*16 = 589824
    if (id >= KT_REG * NT_REG * 32 * 16) return;
    int e    = id & 15;
    int rest = id >> 4;
    int lane = rest & 31;
    int tile = rest >> 5;
    int nt = tile % NT_REG, kt = tile / NT_REG;
    int k  = kt * 32 + (lane >> 4) * 8 + (e & 7) + (e >> 3) * 16;
    int kk = k >> 8, c = k & 255;                     // tap-major
    int n  = nt * 16 + (lane & 15);
    Bh[id] = (_Float16)w_reg[(size_t)n * KRED + (size_t)c * KK + kk];
}

// Pack w_mask [256][256] into B fragments (K=256, N=256).
__global__ void prep_bmask_kernel(const float* __restrict__ w_mask,
                                  _Float16* __restrict__ Bh) {
    int id = blockIdx.x * blockDim.x + threadIdx.x;   // 8*16*32*16 = 65536
    if (id >= KT_MASK * NT_REG * 32 * 16) return;
    int e    = id & 15;
    int rest = id >> 4;
    int lane = rest & 31;
    int tile = rest >> 5;
    int nt = tile % NT_REG, kt = tile / NT_REG;
    int k = kt * 32 + (lane >> 4) * 8 + (e & 7) + (e >> 3) * 16;
    int n = nt * 16 + (lane & 15);
    Bh[id] = (_Float16)w_mask[(size_t)n * CDIM + k];
}

// Pack [w_off(18ch) ; w_mod(9ch) ; pad(5ch)] into B fragments, tap-major k.
__global__ void prep_boffmod_kernel(const float* __restrict__ w_off,
                                    const float* __restrict__ w_mod,
                                    _Float16* __restrict__ Bh) {
    int id = blockIdx.x * blockDim.x + threadIdx.x;   // 72*2*32*16 = 73728
    if (id >= KT_REG * 2 * 32 * 16) return;
    int e    = id & 15;
    int rest = id >> 4;
    int lane = rest & 31;
    int tile = rest >> 5;
    int nt = tile % 2, kt = tile / 2;
    int k  = kt * 32 + (lane >> 4) * 8 + (e & 7) + (e >> 3) * 16;
    int kk = k >> 8, c = k & 255;
    int n  = nt * 16 + (lane & 15);
    float v = 0.f;
    if (n < 18)       v = w_off[(size_t)n * KRED + (size_t)c * KK + kk];
    else if (n < 27)  v = w_mod[(size_t)(n - 18) * KRED + (size_t)c * KK + kk];
    Bh[id] = (_Float16)v;
}

// ---------------------------------------------------------------------------
// Offset + modulator 3x3 conv (implicit im2col GEMM, N=32, fused epilogue)
// 64 threads = 2 waves; wave w owns n-tile w.  Ping-pong LDS A tiles.
// Branch-free fill: clamped base index + f16 zero/one scale per (pixel,tap);
// each thread moves one 16-byte v8h per chunk.
// ---------------------------------------------------------------------------
__global__ void offmod_conv_kernel(const _Float16* __restrict__ xh,
                                   const _Float16* __restrict__ Bh,
                                   const float* __restrict__ b_off,
                                   const float* __restrict__ b_mod,
                                   float* __restrict__ offbuf,   // [NPIX][18]
                                   float* __restrict__ modbuf) { // [NPIX][9]
    __shared__ __align__(16) _Float16 sA[2][16 * 40];
    __shared__ int      sIdx[144];            // clamped NHWC base (halves)
    __shared__ _Float16 sSc[144];             // 1 if in-bounds else 0
    int tile = blockIdx.x;
    int pix0 = tile * 16;
    int b    = tile / TILES_PER_BATCH;
    int tid  = threadIdx.x;
    int wave = tid >> 5, lane = tid & 31;

    for (int i = tid; i < 144; i += 64) {
        int m = i / KK, kk = i - m * KK;
        int pix = pix0 + m;
        int hw  = pix - b * HW;
        int hy = hw >> 6, wx = hw & 63;
        int yy = hy + kk / 3 - 1;
        int xx = wx + kk % 3 - 1;
        bool ok = (yy >= 0 && yy < HDIM && xx >= 0 && xx < WDIM);
        int cy = min(max(yy, 0), HDIM - 1);
        int cx = min(max(xx, 0), WDIM - 1);
        sIdx[i] = (b * HW + cy * WDIM + cx) * CDIM;
        sSc[i]  = ok ? (_Float16)1.f : (_Float16)0.f;
    }
    __syncthreads();

    // One v8h per thread fills the whole 16x32 tile (64 threads).
    auto fill = [&](int buf, int ktn) {
        int kk  = ktn >> 3;
        int c0  = (ktn & 7) * 32;
        int row = tid >> 2;
        int ko  = (tid & 3) * 8;
        int p   = row * KK + kk;
        v8h d = *(const v8h*)(xh + (size_t)sIdx[p] + c0 + ko);
        _Float16 s = sSc[p];
#pragma unroll
        for (int i = 0; i < 8; ++i) d[i] = d[i] * s;
        *(v8h*)(&sA[buf][row * 40 + ko]) = d;
    };

    v8f acc = {};
    fill(0, 0);
    __syncthreads();
    for (int kt = 0; kt < KT_REG; ++kt) {
        int cur = kt & 1;
        v16h a  = load_afrag_lds(sA[cur], lane);
        v16h bf = load_bfrag(Bh, kt, 2, wave, lane);
        if (kt + 1 < KT_REG) fill(1 - cur, kt + 1);
        acc = wmma_f16(a, bf, acc);
        __syncthreads();
    }

    // Epilogue: D element r -> (M = r + 8*(lane>>4), N = wave*16 + lane&15)
    int n  = wave * 16 + (lane & 15);
    int mg = (lane >> 4) * 8;
#pragma unroll
    for (int r = 0; r < 8; ++r) {
        int pix = pix0 + mg + r;
        float v = acc[r];
        if (n < 18) {
            offbuf[(size_t)pix * 18 + n] = v + b_off[n];
        } else if (n < 27) {
            int j = n - 18;
            float z = v + b_mod[j];
            modbuf[(size_t)pix * KK + j] = 2.0f / (1.0f + __expf(-z));
        }
    }
}

// ---------------------------------------------------------------------------
// Deformable conv main GEMM: 16 pixels x 256 out-ch per block, K=2304.
// 256 threads = 8 waves; wave w owns n-tiles 2w, 2w+1.  Ping-pong LDS tiles.
// Fill is loop-free: each thread produces one v2h (4 dword corner loads).
// Output staged in LDS for coalesced fbuf stores; per-tile column sums for SE.
// ---------------------------------------------------------------------------
__global__ void deform_kernel(const _Float16* __restrict__ xh,
                              const float* __restrict__ offbuf,
                              const float* __restrict__ modbuf,
                              const _Float16* __restrict__ Bh,
                              const float* __restrict__ b_reg,
                              float* __restrict__ fbuf,     // [NPIX][256]
                              float* __restrict__ colsum) { // [NTILES][256]
    __shared__ float sw00[144], sw01[144], sw10[144], sw11[144];
    __shared__ int   si00[144], si01[144], si10[144], si11[144];
    __shared__ __align__(16) _Float16 sA[2][16 * 40];
    __shared__ float sOut[16][260];

    int tile = blockIdx.x;
    int pix0 = tile * 16;
    int b    = tile / TILES_PER_BATCH;
    int tid  = threadIdx.x;
    int wave = tid >> 5, lane = tid & 31;

    // Bilinear corner indices / weights (validity-masked, modulator-folded).
    if (tid < 144) {
        int m  = tid / KK, kk = tid - m * KK;
        int pix = pix0 + m;
        int hw  = pix - b * HW;
        int hy = hw >> 6, wx = hw & 63;
        float dy = offbuf[(size_t)pix * 18 + 2 * kk];
        float dx = offbuf[(size_t)pix * 18 + 2 * kk + 1];
        float md = modbuf[(size_t)pix * KK + kk];
        float py = (float)hy + (float)(kk / 3 - 1) + dy;
        float px = (float)wx + (float)(kk % 3 - 1) + dx;
        float y0f = floorf(py), x0f = floorf(px);
        float ly = py - y0f, lx = px - x0f;
        int y0 = (int)y0f, x0 = (int)x0f;
        int y1 = y0 + 1,  x1 = x0 + 1;
        float vy0 = (y0 >= 0 && y0 < HDIM) ? 1.f : 0.f;
        float vy1 = (y1 >= 0 && y1 < HDIM) ? 1.f : 0.f;
        float vx0 = (x0 >= 0 && x0 < WDIM) ? 1.f : 0.f;
        float vx1 = (x1 >= 0 && x1 < WDIM) ? 1.f : 0.f;
        int cy0 = min(max(y0, 0), HDIM - 1), cy1 = min(max(y1, 0), HDIM - 1);
        int cx0 = min(max(x0, 0), WDIM - 1), cx1 = min(max(x1, 0), WDIM - 1);
        int base = b * HW;
        si00[tid] = (base + cy0 * WDIM + cx0) * CDIM;
        si01[tid] = (base + cy0 * WDIM + cx1) * CDIM;
        si10[tid] = (base + cy1 * WDIM + cx0) * CDIM;
        si11[tid] = (base + cy1 * WDIM + cx1) * CDIM;
        sw00[tid] = (1.f - ly) * (1.f - lx) * vy0 * vx0 * md;
        sw01[tid] = (1.f - ly) * lx         * vy0 * vx1 * md;
        sw10[tid] = ly * (1.f - lx)         * vy1 * vx0 * md;
        sw11[tid] = ly * lx                 * vy1 * vx1 * md;
    }
    __syncthreads();

    // One v2h per thread fills the 16x32 tile (256 threads, no loop).
    auto fill = [&](int buf, int ktn) {
        int kk = ktn >> 3;
        int c0 = (ktn & 7) * 32;
        int m  = tid >> 4;
        int k2 = (tid & 15) * 2;
        int p  = m * KK + kk;
        int cb = c0 + k2;
        v2h a00 = *(const v2h*)(xh + (size_t)si00[p] + cb);
        v2h a01 = *(const v2h*)(xh + (size_t)si01[p] + cb);
        v2h a10 = *(const v2h*)(xh + (size_t)si10[p] + cb);
        v2h a11 = *(const v2h*)(xh + (size_t)si11[p] + cb);
        float w0 = sw00[p], w1 = sw01[p], w2 = sw10[p], w3 = sw11[p];
        v2h r;
#pragma unroll
        for (int i = 0; i < 2; ++i)
            r[i] = (_Float16)(w0 * (float)a00[i] + w1 * (float)a01[i]
                            + w2 * (float)a10[i] + w3 * (float)a11[i]);
        *(v2h*)(&sA[buf][m * 40 + k2]) = r;
    };

    int nt0 = wave * 2, nt1 = wave * 2 + 1;
    v8f acc0 = {}, acc1 = {};

    fill(0, 0);
    __syncthreads();
    for (int kt = 0; kt < KT_REG; ++kt) {
        int cur = kt & 1;
        v16h a  = load_afrag_lds(sA[cur], lane);
        v16h b0 = load_bfrag(Bh, kt, NT_REG, nt0, lane);
        v16h b1 = load_bfrag(Bh, kt, NT_REG, nt1, lane);
        if (kt + 1 < KT_REG) fill(1 - cur, kt + 1);
        acc0 = wmma_f16(a, b0, acc0);
        acc1 = wmma_f16(a, b1, acc1);
        __syncthreads();
    }

    // Epilogue: bias, per-tile column sums, stage tile in LDS.
    int ncol = lane & 15;
    int mg   = (lane >> 4) * 8;
#pragma unroll
    for (int t = 0; t < 2; ++t) {
        v8f acc = t ? acc1 : acc0;
        int n = (t ? nt1 : nt0) * 16 + ncol;
        float bias = b_reg[n];
        float csum = 0.f;
#pragma unroll
        for (int r = 0; r < 8; ++r) {
            float fv = acc[r] + bias;
            sOut[mg + r][n] = fv;
            csum += fv;
        }
        csum += __shfl_xor(csum, 16);   // combine the two 8-row halves
        if (lane < 16) colsum[(size_t)tile * CDIM + n] = csum;
    }
    __syncthreads();
    // Coalesced row-major stores of the 16x256 tile.
    for (int i = tid; i < 16 * CDIM; i += 256) {
        int m = i >> 8, c = i & 255;
        fbuf[(size_t)(pix0 + m) * CDIM + c] = sOut[m][c];
    }
}

// ---------------------------------------------------------------------------
// SE: deterministic mean reduction + tiny MLP.
// ---------------------------------------------------------------------------
__global__ void colsum_reduce_kernel(const float* __restrict__ colsum,
                                     float* __restrict__ meanbuf) {
    int b = blockIdx.x, c = threadIdx.x;
    float s = 0.f;
    for (int t = 0; t < TILES_PER_BATCH; ++t)
        s += colsum[((size_t)(b * TILES_PER_BATCH + t)) * CDIM + c];
    meanbuf[b * CDIM + c] = s * (1.0f / (float)HW);
}

__global__ void se_mlp_kernel(const float* __restrict__ meanbuf,
                              const float* __restrict__ w1,  // [64][256]
                              const float* __restrict__ b1,
                              const float* __restrict__ w2,  // [256][64]
                              const float* __restrict__ b2,
                              float* __restrict__ sbuf) {    // [B][256]
    __shared__ float sm[CDIM];
    __shared__ float sh[64];
    int b = blockIdx.x, tid = threadIdx.x;
    sm[tid] = meanbuf[b * CDIM + tid];
    __syncthreads();
    if (tid < 64) {
        float h = b1[tid];
        for (int c = 0; c < CDIM; ++c) h += sm[c] * w1[tid * CDIM + c];
        sh[tid] = fmaxf(h, 0.f);
    }
    __syncthreads();
    float z = b2[tid];
    for (int j = 0; j < 64; ++j) z += sh[j] * w2[tid * 64 + j];
    sbuf[b * CDIM + tid] = 1.0f / (1.0f + __expf(-z));
}

// ---------------------------------------------------------------------------
// Final: fused 1x1 mask conv (WMMA, K=256) + hardsigmoid + out = f*s*m + x.
// 256 threads = 8 waves; wave w owns n-tiles 2w, 2w+1; A straight from xh.
// fbuf pre-loaded coalesced into LDS; out/x handled in NCHW-friendly order.
// ---------------------------------------------------------------------------
__global__ void final_kernel(const _Float16* __restrict__ xh,
                             const _Float16* __restrict__ Bh,
                             const float* __restrict__ b_mask,
                             const float* __restrict__ fbuf,
                             const float* __restrict__ sbuf,
                             const float* __restrict__ x,
                             float* __restrict__ out) {
    __shared__ float sF[16][260];
    int tile = blockIdx.x;
    int pix0 = tile * 16;
    int b    = tile / TILES_PER_BATCH;
    int tid  = threadIdx.x;
    int lane = tid & 31;
    int wave = tid >> 5;
    int nt0 = wave * 2, nt1 = wave * 2 + 1;

    // Coalesced pre-load of the f tile.
    for (int i = tid; i < 16 * CDIM; i += 256) {
        int m = i >> 8, c = i & 255;
        sF[m][c] = fbuf[(size_t)(pix0 + m) * CDIM + c];
    }

    int row = lane & 15;
    int kb  = (lane >> 4) * 8;
    const _Float16* arow = xh + (size_t)(pix0 + row) * CDIM;

    v8f acc0 = {}, acc1 = {};
#pragma unroll
    for (int kt = 0; kt < KT_MASK; ++kt) {
        v8h lo = *(const v8h*)(arow + kt * 32 + kb);
        v8h hi = *(const v8h*)(arow + kt * 32 + kb + 16);
        v16h a;
#pragma unroll
        for (int i = 0; i < 8; ++i) { a[i] = lo[i]; a[i + 8] = hi[i]; }
        acc0 = wmma_f16(a, load_bfrag(Bh, kt, NT_REG, nt0, lane), acc0);
        acc1 = wmma_f16(a, load_bfrag(Bh, kt, NT_REG, nt1, lane), acc1);
    }
    __syncthreads();   // sF ready; wmma + preload complete

    // Scale staged f by SE * hardsigmoid(mask) in LDS (each cell owned by
    // exactly one lane).
    int ncol = lane & 15;
    int mg   = (lane >> 4) * 8;
#pragma unroll
    for (int t = 0; t < 2; ++t) {
        v8f acc = t ? acc1 : acc0;
        int n = (t ? nt1 : nt0) * 16 + ncol;
        float mb = b_mask[n];
        float sv = sbuf[b * CDIM + n];
#pragma unroll
        for (int r = 0; r < 8; ++r) {
            float mval = fminf(fmaxf((acc[r] + mb) * (1.0f / 6.0f) + 0.5f, 0.f), 1.f);
            sF[mg + r][n] *= sv * mval;
        }
    }
    __syncthreads();

    // out = staged + x, written in NCHW order (16 consecutive hw per channel).
    int hw0 = pix0 - b * HW;
    for (int i = tid; i < 16 * CDIM; i += 256) {
        int c = i >> 4, ml = i & 15;
        size_t idx = ((size_t)(b * CDIM + c)) * HW + hw0 + ml;
        out[idx] = sF[ml][c] + x[idx];
    }
}

// ---------------------------------------------------------------------------
// Host launcher
// ---------------------------------------------------------------------------
extern "C" void kernel_launch(void* const* d_in, const int* in_sizes, int n_in,
                              void* d_out, int out_size, void* d_ws, size_t ws_size,
                              hipStream_t stream) {
    (void)in_sizes; (void)n_in; (void)out_size; (void)ws_size;
    const float* x      = (const float*)d_in[0];
    const float* w_off  = (const float*)d_in[1];
    const float* b_off  = (const float*)d_in[2];
    const float* w_mod  = (const float*)d_in[3];
    const float* b_mod  = (const float*)d_in[4];
    const float* w_reg  = (const float*)d_in[5];
    const float* b_reg  = (const float*)d_in[6];
    const float* w_se1  = (const float*)d_in[7];
    const float* b_se1  = (const float*)d_in[8];
    const float* w_se2  = (const float*)d_in[9];
    const float* b_se2  = (const float*)d_in[10];
    const float* w_mask = (const float*)d_in[11];
    const float* b_mask = (const float*)d_in[12];
    float* out = (float*)d_out;

    // Workspace carve-out (256-byte aligned slabs).
    char* ws = (char*)d_ws;
    size_t off = 0;
    auto alloc = [&](size_t bytes) {
        char* p = ws + off;
        off += (bytes + 255) & ~(size_t)255;
        return p;
    };
    _Float16* xh       = (_Float16*)alloc((size_t)NPIX * CDIM * 2);           // 16.8 MB
    _Float16* Bh_reg   = (_Float16*)alloc((size_t)KT_REG * NT_REG * 512 * 2); // 1.18 MB
    _Float16* Bh_mask  = (_Float16*)alloc((size_t)KT_MASK * NT_REG * 512 * 2);
    _Float16* Bh_om    = (_Float16*)alloc((size_t)KT_REG * 2 * 512 * 2);
    float*    offbuf   = (float*)alloc((size_t)NPIX * 18 * 4);
    float*    modbuf   = (float*)alloc((size_t)NPIX * KK * 4);
    float*    fbuf     = (float*)alloc((size_t)NPIX * CDIM * 4);              // 33.6 MB
    float*    colsum   = (float*)alloc((size_t)NTILES * CDIM * 4);
    float*    meanbuf  = (float*)alloc((size_t)BATCH * CDIM * 4);
    float*    sbuf     = (float*)alloc((size_t)BATCH * CDIM * 4);

    // 1) layout conversions
    prep_xh_kernel<<<BATCH * CDIM, 256, 0, stream>>>(x, xh);
    prep_breg_kernel<<<(KT_REG * NT_REG * 512 + 255) / 256, 256, 0, stream>>>(w_reg, Bh_reg);
    prep_bmask_kernel<<<(KT_MASK * NT_REG * 512 + 255) / 256, 256, 0, stream>>>(w_mask, Bh_mask);
    prep_boffmod_kernel<<<(KT_REG * 2 * 512 + 255) / 256, 256, 0, stream>>>(w_off, w_mod, Bh_om);

    // 2) offset + modulator convs (WMMA im2col GEMM)
    offmod_conv_kernel<<<NTILES, 64, 0, stream>>>(xh, Bh_om, b_off, b_mod, offbuf, modbuf);

    // 3) deformable conv main contraction (WMMA) + SE partial sums
    deform_kernel<<<NTILES, 256, 0, stream>>>(xh, offbuf, modbuf, Bh_reg, b_reg, fbuf, colsum);

    // 4) SE mean + MLP
    colsum_reduce_kernel<<<BATCH, 256, 0, stream>>>(colsum, meanbuf);
    se_mlp_kernel<<<BATCH, 256, 0, stream>>>(meanbuf, w_se1, b_se1, w_se2, b_se2, sbuf);

    // 5) fused mask conv (WMMA) + combine
    final_kernel<<<NTILES, 256, 0, stream>>>(xh, Bh_mask, b_mask, fbuf, sbuf, x, out);
}
